// decoder_16673063043336
// MI455X (gfx1250) — matrix operand order repaired and checked by reading, not compiled, not run on hardware
//
#include <hip/hip_runtime.h>
#include <math.h>
#include <stdint.h>

typedef __attribute__((ext_vector_type(16))) __bf16 v16bf;
typedef __attribute__((ext_vector_type(8)))  __bf16 v8bf;
typedef __attribute__((ext_vector_type(8)))  float  v8f;
typedef __attribute__((ext_vector_type(4)))  unsigned int v4u;
typedef __attribute__((ext_vector_type(8)))  int v8i_;
typedef __attribute__((ext_vector_type(4)))  int v4i_;

namespace {
constexpr int Vv = 32000, Ee = 512, Hh = 1024, Bb = 32, Ss = 64, Tt = 64;
}

// ---------------------------------------------------------------------------
// TDM: issue a 2D tensor tile load (bf16 elements) into LDS.
// Descriptor layout per CDNA5 ISA 8.3/8.4 (D# group0 128b, group1 256b).
// ---------------------------------------------------------------------------
__device__ __forceinline__ unsigned lds_offset(const void* p) {
  // LDS aperture: flat addr[31:0] == LDS byte offset (ISA 10.2 aperture table)
  return (unsigned)(uintptr_t)p;
}

__device__ __forceinline__ void tdm_load_2d_bf16(unsigned lds_off,
                                                 const __bf16* gptr,
                                                 unsigned tensor_d0,  // row len (elems)
                                                 unsigned tensor_d1,  // #rows
                                                 unsigned stride_d0,  // row pitch (elems)
                                                 unsigned tile_d0,    // tile width
                                                 unsigned tile_d1) {  // tile height
  unsigned long long ga = (unsigned long long)(uintptr_t)gptr;
  v4u g0;
  g0[0] = 1u;                                   // count=1, user descriptor
  g0[1] = lds_off;                              // lds_addr [63:32]
  g0[2] = (unsigned)ga;                         // global_addr low
  g0[3] = (unsigned)((ga >> 32) & 0x01FFFFFFu)  // global_addr[56:32]
          | (2u << 30);                         // type=2 ("image")
  v8i_ g1;
  g1[0] = (int)(1u << 16);                      // wg_mask=0, data_size=1 (2 bytes)
  g1[1] = (int)((tensor_d0 & 0xFFFFu) << 16);   // tensor_dim0[15:0] @ [63:48]
  g1[2] = (int)((tensor_d0 >> 16) | ((tensor_d1 & 0xFFFFu) << 16));
  g1[3] = (int)((tensor_d1 >> 16) | (tile_d0 << 16));   // tile_dim0 @ [127:112]
  g1[4] = (int)tile_d1;                         // tile_dim1; tile_dim2=0
  g1[5] = (int)stride_d0;                       // tensor_dim0_stride[31:0]
  g1[6] = 0;                                    // stride[47:32]=0, dim1_stride lo=0
  g1[7] = 0;
  v4i_ z4 = {0, 0, 0, 0};
  v8i_ z8 = {0, 0, 0, 0, 0, 0, 0, 0};
  // 6-arg form (clang-23 / therock headers): (g0, g1, g2, g3, g4, cpol)
  __builtin_amdgcn_tensor_load_to_lds(g0, g1, z4, z4, z8, 0);
}

// ---------------------------------------------------------------------------
// TDM + LDS staged bf16 GEMM: C[M,N](f32) = Ab[M,K](bf16) @ Wb[K,N](bf16) + bias
// 256 threads/block, macro tile 32(M) x 64(N), K-step 32.
// remap!=0: row r = t*B+b is written to output row b*T+t (out projection).
// ---------------------------------------------------------------------------
__global__ __launch_bounds__(256)
void gemm_bf16_tdm_kernel(const __bf16* __restrict__ Ab,
                          const __bf16* __restrict__ Wb,
                          const float*  __restrict__ bias,
                          float* __restrict__ C,
                          int M, int N, int K, int remap) {
  __shared__ __bf16 As[32][32];   // M x K tile
  __shared__ __bf16 Bs[32][64];   // K x N tile
  int nblk = N >> 6;
  int mb = blockIdx.x / nblk;
  int nb = blockIdx.x % nblk;
  int mbase = mb << 5, nbase0 = nb << 6;
  int wave = threadIdx.x >> 5, lane = threadIdx.x & 31;

  int mt = (wave >> 2) << 4;      // 0 / 16
  int nt = (wave & 3) << 4;       // 0 / 16 / 32 / 48
  v8f acc = {};

  for (int k0 = 0; k0 < K; k0 += 32) {
    __syncthreads();              // protect LDS from previous iteration readers
    if (wave == 0) {
      tdm_load_2d_bf16(lds_offset(&As[0][0]), Ab + (size_t)mbase * K + k0,
                       (unsigned)K, (unsigned)M, (unsigned)K, 32u, 32u);
      tdm_load_2d_bf16(lds_offset(&Bs[0][0]), Wb + (size_t)k0 * N + nbase0,
                       (unsigned)N, (unsigned)K, (unsigned)N, 64u, 32u);
      __builtin_amdgcn_s_wait_tensorcnt(0);
    }
    __syncthreads();

    // A fragment (ISA 16-bit A 16x32 layout): two 16B LDS reads per lane
    int m = mt + (lane & 15);
    int kh = ((lane >> 4) & 1) << 3;
    v8bf lo = *(const v8bf*)&As[m][kh];
    v8bf hi = *(const v8bf*)&As[m][kh + 16];
    v16bf a = __builtin_shufflevector(lo, hi, 0, 1, 2, 3, 4, 5, 6, 7,
                                      8, 9, 10, 11, 12, 13, 14, 15);
    // B fragment: lanes 0-15 K=0..7,16..23, lanes 16-31 K=8..15,24..31, N = lane&15
    int n = nt + (lane & 15);
    int kb = ((lane >> 4) & 1) << 3;
    v16bf b;
#pragma unroll
    for (int j = 0; j < 16; ++j) b[j] = Bs[kb + ((j & 8) << 1) + (j & 7)][n];

    acc = __builtin_amdgcn_wmma_f32_16x16x32_bf16(false, a, false, b,
                                                  (short)0, acc, false, false);
  }

  int n = nbase0 + nt + (lane & 15);
  int mo = ((lane >> 4) & 1) << 3;
  float bv = bias[n];
#pragma unroll
  for (int i = 0; i < 8; ++i) {
    int r = mbase + mt + mo + i;
    int orow = remap ? ((r & 31) * Tt + (r >> 5)) : r;   // (t*B+b) -> b*T+t
    C[(size_t)orow * N + n] = acc[i] + bv;
  }
}

// ---------------------------------------------------------------------------
// Direct-gather bf16 fragment loaders from fp32 memory (used by the small,
// latency-bound recurrence cell where staging overhead isn't worth it).
// ---------------------------------------------------------------------------
__device__ __forceinline__ v16bf load_a_frag(const float* __restrict__ A, int lda,
                                             int mbase, int k0) {
  int lane = threadIdx.x & 31;
  int m = mbase + (lane & 15);
  int kh = ((lane >> 4) & 1) << 3;
  const float* row = A + (size_t)m * lda + k0 + kh;
  v16bf a;
#pragma unroll
  for (int j = 0; j < 16; ++j) {
    int k = ((j & 8) << 1) + (j & 7);
    a[j] = (__bf16)row[k];
  }
  return a;
}

__device__ __forceinline__ v16bf load_b_frag(const float* __restrict__ W, int ldw,
                                             int k0, int nbase) {
  int lane = threadIdx.x & 31;
  int n = nbase + (lane & 15);
  int kh = ((lane >> 4) & 1) << 3;
  const float* p = W + (size_t)(k0 + kh) * ldw + n;
  v16bf b;
#pragma unroll
  for (int j = 0; j < 16; ++j) b[j] = (__bf16)p[(size_t)(((j & 8) << 1) + (j & 7)) * ldw];
  return b;
}

// ---- Bahdanau attention, one timestep. One block per batch element. --------
__global__ __launch_bounds__(1024)
void attention_step_kernel(const float* __restrict__ enc_proj, // [B,S,H]
                           const float* __restrict__ enc_h,    // [B,S,2H]
                           const float* __restrict__ s_prev,   // [B,H]
                           const float* __restrict__ Wprev,    // [H,H]
                           const float* __restrict__ bprev,    // [H]
                           const float* __restrict__ wlin,     // [H]
                           const float* __restrict__ blin,     // [1]
                           float* __restrict__ ctx) {          // [B,2H]
  __shared__ float s_s[Hh];
  __shared__ float s_q[Hh];
  __shared__ float s_sc[Ss];
  __shared__ float s_red[2];
  int b = blockIdx.x;
  int tid = threadIdx.x;

  s_s[tid] = s_prev[(size_t)b * Hh + tid];
  __syncthreads();

  float q = bprev[tid];
  for (int k = 0; k < Hh; ++k) q = fmaf(s_s[k], Wprev[(size_t)k * Hh + tid], q);
  s_q[tid] = q;
  __syncthreads();

  int wave = tid >> 5, lane = tid & 31;
  const float* ep = enc_proj + (size_t)b * Ss * Hh;
  for (int s = wave; s < Ss; s += 32) {
    const float* row = ep + (size_t)s * Hh;
    float sum = 0.f;
    for (int h = lane; h < Hh; h += 32)
      sum += tanhf(row[h] + s_q[h]) * wlin[h];
    for (int off = 16; off > 0; off >>= 1)
      sum += __shfl_down(sum, off, 32);
    if (lane == 0) s_sc[s] = sum + blin[0];
  }
  __syncthreads();

  if (tid == 0) {
    float mx = s_sc[0];
    for (int s = 1; s < Ss; ++s) mx = fmaxf(mx, s_sc[s]);
    s_red[0] = mx;
  }
  __syncthreads();
  if (tid < Ss) s_sc[tid] = expf(s_sc[tid] - s_red[0]);
  __syncthreads();
  if (tid == 0) {
    float sum = 0.f;
    for (int s = 0; s < Ss; ++s) sum += s_sc[s];
    s_red[1] = 1.f / sum;
  }
  __syncthreads();

  float inv = s_red[1];
  const float* eh = enc_h + (size_t)b * Ss * 2 * Hh;
  float a0 = 0.f, a1 = 0.f;
  for (int s = 0; s < Ss; ++s) {
    float al = s_sc[s] * inv;
    a0 = fmaf(al, eh[(size_t)s * 2 * Hh + tid], a0);
    a1 = fmaf(al, eh[(size_t)s * 2 * Hh + Hh + tid], a1);
  }
  ctx[(size_t)b * 2 * Hh + tid] = a0;
  ctx[(size_t)b * 2 * Hh + Hh + tid] = a1;
}

// ---- Fused GRU-like cell, one timestep. 64 blocks x 4 waves. ---------------
__global__ __launch_bounds__(128)
void cell_step_kernel(const float* __restrict__ x_t,     // [B,E]
                      const float* __restrict__ s_prev,  // [B,H]
                      const float* __restrict__ ctx,     // [B,2H]
                      const float* __restrict__ Wi,  const float* __restrict__ bi,
                      const float* __restrict__ Wh,  const float* __restrict__ bh,
                      const float* __restrict__ Wc,  const float* __restrict__ bc,
                      const float* __restrict__ Wi2, const float* __restrict__ bi2,
                      const float* __restrict__ Wh2, const float* __restrict__ bh2,
                      const float* __restrict__ Wc2, const float* __restrict__ bc2,
                      float* __restrict__ s_new) {       // [B,H]
  __shared__ float sm[2][32][16];
  int wave = threadIdx.x >> 5;
  int matsel = wave >> 1;
  int mbase = (wave & 1) << 4;
  int nbase = blockIdx.x << 4;

  const float* Wia = matsel ? Wi2 : Wi;
  const float* Wha = matsel ? Wh2 : Wh;
  const float* Wca = matsel ? Wc2 : Wc;
  int ldw = matsel ? Hh : 2 * Hh;

  v8f acc = {};
  for (int k0 = 0; k0 < Ee; k0 += 32) {
    v16bf a = load_a_frag(x_t, Ee, mbase, k0);
    v16bf w = load_b_frag(Wia, ldw, k0, nbase);
    acc = __builtin_amdgcn_wmma_f32_16x16x32_bf16(false, a, false, w,
                                                  (short)0, acc, false, false);
  }
  for (int k0 = 0; k0 < Hh; k0 += 32) {
    v16bf a = load_a_frag(s_prev, Hh, mbase, k0);
    v16bf w = load_b_frag(Wha, ldw, k0, nbase);
    acc = __builtin_amdgcn_wmma_f32_16x16x32_bf16(false, a, false, w,
                                                  (short)0, acc, false, false);
  }
  for (int k0 = 0; k0 < 2 * Hh; k0 += 32) {
    v16bf a = load_a_frag(ctx, 2 * Hh, mbase, k0);
    v16bf w = load_b_frag(Wca, ldw, k0, nbase);
    acc = __builtin_amdgcn_wmma_f32_16x16x32_bf16(false, a, false, w,
                                                  (short)0, acc, false, false);
  }

  int lane = threadIdx.x & 31;
  int n = lane & 15;
  int mo = ((lane >> 4) & 1) << 3;
  int gn = nbase + n;
  float bv = matsel ? (bi2[gn] + bh2[gn] + bc2[gn])
                    : (bi[gn] + bh[gn] + bc[gn]);
#pragma unroll
  for (int i = 0; i < 8; ++i)
    sm[matsel][mbase + mo + i][n] = acc[i] + bv;
  __syncthreads();

  for (int o = threadIdx.x; o < 32 * 16; o += 128) {
    int m = o >> 4, nn = o & 15;
    int col = nbase + nn;
    float g = sm[0][m][nn];
    float c = sm[1][m][nn];
    float r = 1.f / (1.f + expf(-g));
    float sp = s_prev[(size_t)m * Hh + col];
    s_new[(size_t)m * Hh + col] = (1.f - r) * sp + r * tanhf(c);
  }
}

// ---- Elementwise helpers ---------------------------------------------------
__global__ void embed_gather_kernel(const int* __restrict__ target,
                                    const float* __restrict__ embed,
                                    float* __restrict__ xs) {
  int idx = blockIdx.x * blockDim.x + threadIdx.x;
  int total = Tt * Bb * Ee;
  if (idx >= total) return;
  int e = idx % Ee;
  int rb = idx / Ee;
  int b = rb % Bb;
  int t = rb / Bb;
  int tok = target[b * Tt + t];
  xs[idx] = embed[(size_t)tok * Ee + e];
}

__global__ void f32_to_bf16_kernel(const float* __restrict__ src,
                                   __bf16* __restrict__ dst, int n) {
  int i = blockIdx.x * blockDim.x + threadIdx.x;
  if (i < n) dst[i] = (__bf16)src[i];
}

__global__ void copy_final_kernel(const float* __restrict__ src,
                                  float* __restrict__ dst, int n) {
  int i = blockIdx.x * blockDim.x + threadIdx.x;
  if (i < n) dst[i] = src[i];
}

// ---------------------------------------------------------------------------
extern "C" void kernel_launch(void* const* d_in, const int* in_sizes, int n_in,
                              void* d_out, int out_size, void* d_ws, size_t ws_size,
                              hipStream_t stream) {
  const float* enc_h     = (const float*)d_in[0];
  const float* prev_s    = (const float*)d_in[1];
  const int*   target    = (const int*)d_in[2];
  const float* embed     = (const float*)d_in[3];
  const float* att_Wenc  = (const float*)d_in[4];
  const float* att_benc  = (const float*)d_in[5];
  const float* att_Wprev = (const float*)d_in[6];
  const float* att_bprev = (const float*)d_in[7];
  const float* att_wlin  = (const float*)d_in[8];
  const float* att_blin  = (const float*)d_in[9];
  const float* dc_Wi  = (const float*)d_in[10];
  const float* dc_bi  = (const float*)d_in[11];
  const float* dc_Wh  = (const float*)d_in[12];
  const float* dc_bh  = (const float*)d_in[13];
  const float* dc_Wc  = (const float*)d_in[14];
  const float* dc_bc  = (const float*)d_in[15];
  const float* dc_Wi2 = (const float*)d_in[16];
  const float* dc_bi2 = (const float*)d_in[17];
  const float* dc_Wh2 = (const float*)d_in[18];
  const float* dc_bh2 = (const float*)d_in[19];
  const float* dc_Wc2 = (const float*)d_in[20];
  const float* dc_bc2 = (const float*)d_in[21];
  const float* out_W  = (const float*)d_in[22];
  const float* out_b  = (const float*)d_in[23];
  float* out = (float*)d_out;

  // workspace layout: f32 scratch then bf16 staging copies (~104 MB total)
  float* encp  = (float*)d_ws;                       // B*S*H  = 2,097,152 f32
  float* xs    = encp  + (size_t)Bb * Ss * Hh;       // T*B*E  = 1,048,576 f32
  float* dec_h = xs    + (size_t)Tt * Bb * Ee;       // T*B*H  = 2,097,152 f32
  float* ctx   = dec_h + (size_t)Tt * Bb * Hh;       // B*2H   =    65,536 f32
  __bf16* ehb   = (__bf16*)(ctx + (size_t)Bb * 2 * Hh);   // enc_h   4,194,304
  __bf16* wencb = ehb   + (size_t)Bb * Ss * 2 * Hh;       // Wenc    2,097,152
  __bf16* dhb   = wencb + (size_t)2 * Hh * Hh;            // dec_h   2,097,152
  __bf16* owb   = dhb   + (size_t)Tt * Bb * Hh;           // out_W  32,768,000

  // 1. xs[t,b,:] = embed[target[b,t]]
  {
    int total = Tt * Bb * Ee;
    embed_gather_kernel<<<(total + 255) / 256, 256, 0, stream>>>(target, embed, xs);
  }
  // 2. bf16 staging of enc_h and att_Wenc, then enc_proj GEMM (TDM path)
  {
    int ne = Bb * Ss * 2 * Hh;
    f32_to_bf16_kernel<<<(ne + 255) / 256, 256, 0, stream>>>(enc_h, ehb, ne);
    int nw = 2 * Hh * Hh;
    f32_to_bf16_kernel<<<(nw + 255) / 256, 256, 0, stream>>>(att_Wenc, wencb, nw);
    int grid = ((Bb * Ss) / 32) * (Hh / 64);           // 64 * 16
    gemm_bf16_tdm_kernel<<<grid, 256, 0, stream>>>(
        ehb, wencb, att_benc, encp, Bb * Ss, Hh, 2 * Hh, 0);
  }
  // 3. sequential recurrence over T
  for (int t = 0; t < Tt; ++t) {
    const float* sp = (t == 0) ? prev_s : dec_h + (size_t)(t - 1) * Bb * Hh;
    attention_step_kernel<<<Bb, Hh, 0, stream>>>(
        encp, enc_h, sp, att_Wprev, att_bprev, att_wlin, att_blin, ctx);
    cell_step_kernel<<<Hh / 16, 128, 0, stream>>>(
        xs + (size_t)t * Bb * Ee, sp, ctx,
        dc_Wi, dc_bi, dc_Wh, dc_bh, dc_Wc, dc_bc,
        dc_Wi2, dc_bi2, dc_Wh2, dc_bh2, dc_Wc2, dc_bc2,
        dec_h + (size_t)t * Bb * Hh);
  }
  // 4. bf16 staging of dec_h and out_W, then logits GEMM (TDM path)
  {
    int nd = Tt * Bb * Hh;
    f32_to_bf16_kernel<<<(nd + 255) / 256, 256, 0, stream>>>(dec_h, dhb, nd);
    int nw = Hh * Vv;
    f32_to_bf16_kernel<<<(nw + 255) / 256, 256, 0, stream>>>(out_W, owb, nw);
    int grid = ((Tt * Bb) / 32) * (Vv / 64);           // 64 * 500
    gemm_bf16_tdm_kernel<<<grid, 256, 0, stream>>>(
        dhb, owb, out_b, out, Tt * Bb, Vv, Hh, 1);
  }
  // 5. final_s = dec_h[T-1]
  copy_final_kernel<<<(Bb * Hh + 255) / 256, 256, 0, stream>>>(
      dec_h + (size_t)(Tt - 1) * Bb * Hh, out + (size_t)Bb * Tt * Vv, Bb * Hh);
}
